// BidirectionalMamba_15925738734297
// MI455X (gfx1250) — compile-verified, matrix operand order checked by reading, not run on hardware
//
#include <hip/hip_runtime.h>
#include <hip/hip_bf16.h>
#include <stdint.h>

// ---------------- problem constants (match reference) ----------------
constexpr int DM = 512;
constexpr int DI = 512;
constexpr int NS = 16;     // state dim N
constexpr int KC = 4;      // conv kernel
constexpr int RR = 32;     // dt rank
constexpr int BB = 4;
constexpr int LL = 1024;
constexpr int MROWS = BB * LL;          // 4096
constexpr int NXZ = 2 * DI;             // 1024
constexpr int NDBC = RR + 2 * NS;       // 64
constexpr int KDT = 64;                 // W_dt K padded 32 -> 64 (zeros)

typedef __attribute__((ext_vector_type(16))) _Float16 v16h;
typedef __attribute__((ext_vector_type(4)))  _Float16 v4h;
typedef __attribute__((ext_vector_type(8)))  float    v8f;

// -------------------------------------------------------------------------
// Block-level GEMM on pre-converted f16 operands.
// 128 threads = 4 waves; block computes a 16(M) x 64(N) slab, wave w owns
// columns [w*16, w*16+16). K-step 64 => two v_wmma per barrier pair.
// A: row-major f16, Abase + r*astride + k (astride in halfs, may be <0: flip)
// B: PRE-TRANSPOSED f16 (BT[n][k], row length bK halfs) -> LDS slabs are
//    contiguous 128B rows for both operands; staged with
//    global_load_async_to_lds_b128 (ASYNCcnt) -- no VGPR round trip.
// Requires blockDim.x == 128; all lanes active (EXEC all ones for WMMA).
// -------------------------------------------------------------------------
__device__ __forceinline__ v8f wmma_gemm_16x64_f16(const _Float16* Abase, long astride,
                                                   const _Float16* BTblk, long bK, int Ktot)
{
    __shared__ _Float16 As[16 * 64];    // [row][k]   2 KB
    __shared__ _Float16 Bs[64 * 64];    // [col][k]   8 KB

    const int t     = threadIdx.x;      // 0..127
    const int lane  = t & 31;
    const int wv    = t >> 5;
    const int half  = lane >> 4;        // 0: lanes 0-15, 1: lanes 16-31
    const int row16 = lane & 15;

    const uint32_t asB = (uint32_t)(uintptr_t)(void*)As;
    const uint32_t bsB = (uint32_t)(uintptr_t)(void*)Bs;

    // A: 16 rows x 64 halfs = 2KB = 128 x 16B chunks; thread t -> one chunk
    const int ar  = t >> 3;             // row 0..15
    const int akq = t & 7;              // 8-half group within row
    const uint32_t ldsA = asB + (uint32_t)(ar * 64 + 8 * akq) * 2;
    const _Float16* gA  = Abase + (long)ar * astride + 8 * akq;

    v8f acc = {};
    for (int k0 = 0; k0 < Ktot; k0 += 64) {
        // ---- async stage A tile ----
        asm volatile("global_load_async_to_lds_b128 %0, %1, off"
                     :: "v"(ldsA), "v"(gA) : "memory");
        // ---- async stage B slab: 64 cols x 64 halfs = 8KB = 512 chunks ----
        #pragma unroll
        for (int i = 0; i < 4; ++i) {
            int c   = t + 128 * i;      // chunk id 0..511
            int col = c >> 3;           // 0..63
            int kq  = c & 7;
            uint32_t ldsB = bsB + (uint32_t)(col * 64 + 8 * kq) * 2;
            const _Float16* gB = BTblk + (long)col * bK + (k0 + 8 * kq);
            asm volatile("global_load_async_to_lds_b128 %0, %1, off"
                         :: "v"(ldsB), "v"(gB) : "memory");
        }
        gA += 64;
        asm volatile("s_wait_asynccnt 0" ::: "memory");
        __syncthreads();

        // Per-lane WMMA fragments (ISA 7.12.2): lanes 0-15 K-base 0, lanes
        // 16-31 K-base 8; elems 0..7 -> kb..kb+7, elems 8..15 -> kb+16..kb+23.
        const _Float16* ap = &As[row16 * 64];
        const _Float16* bp = &Bs[(wv * 16 + row16) * 64];
        const int kb = half * 8;
        #pragma unroll
        for (int s = 0; s < 2; ++s) {
            v16h a, b;
            #pragma unroll
            for (int v = 0; v < 8; ++v) {
                int kk = 32 * s + ((v < 4) ? (kb + 2 * v) : (kb + 16 + 2 * (v - 4)));
                a[2 * v]     = ap[kk];
                a[2 * v + 1] = ap[kk + 1];
                b[2 * v]     = bp[kk];
                b[2 * v + 1] = bp[kk + 1];
            }
            acc = __builtin_amdgcn_wmma_f32_16x16x32_f16(
                false, a, false, b, (short)0, acc, false, false);
        }
        __syncthreads();
    }
    return acc;
}

// C/D layout: vgpr r, lanes 0-15 -> M=r, lanes 16-31 -> M=8+r; N = lane&15.

// ---------------- conversion kernels ----------------
// x (f32) -> x16 (f16), 4 elements / thread
__global__ void k_cvt_x(const float* __restrict__ src, _Float16* __restrict__ dst, long n4)
{
    long i = (long)blockIdx.x * blockDim.x + threadIdx.x;
    if (i >= n4) return;
    const float4 v = *(const float4*)(src + 4 * i);
    v4h h4 = { (_Float16)v.x, (_Float16)v.y, (_Float16)v.z, (_Float16)v.w };
    *(v4h*)(dst + 4 * i) = h4;
}

// W (KxN f32, row-major) -> WT (NxKpad f16), zero-padded K
__global__ void k_cvt_wT(const float* __restrict__ W, _Float16* __restrict__ WT,
                         int K, int N, int Kpad)
{
    long e = (long)blockIdx.x * blockDim.x + threadIdx.x;
    if (e >= (long)N * Kpad) return;
    int n = (int)(e / Kpad);
    int k = (int)(e % Kpad);
    float v = (k < K) ? W[(long)k * N + n] : 0.f;
    WT[e] = (_Float16)v;
}

// ---------------- kernel 1: xz = x_dir @ W_in  (4096 x 512 x 1024) ----------------
__global__ void k_gemm_xz(const _Float16* __restrict__ x16, const _Float16* __restrict__ WinT,
                          float* __restrict__ xz, int flip)
{
    const int tiles_m = MROWS / 16;              // 256
    const int tm = blockIdx.x % tiles_m;
    const int tg = blockIdx.x / tiles_m;         // 0..15 (groups of 64 cols)
    const int m0 = tm * 16;
    const int bbq = m0 / LL;
    const int l0  = m0 % LL;

    const _Float16* Abase;
    long astride;
    if (flip) { Abase = x16 + ((long)bbq * LL + (LL - 1 - l0)) * DM; astride = -DM; }
    else      { Abase = x16 + (long)m0 * DM;                         astride =  DM; }

    v8f acc = wmma_gemm_16x64_f16(Abase, astride, WinT + (long)tg * 64 * DM, DM, DM);

    const int lane = threadIdx.x & 31;
    const int wv   = threadIdx.x >> 5;
    const int half = lane >> 4, col = lane & 15;
    const int nbase = tg * 64 + wv * 16 + col;
    #pragma unroll
    for (int r = 0; r < 8; ++r) {
        int mr = m0 + r + 8 * half;
        xz[(long)mr * NXZ + nbase] = acc[r];
    }
}

// ---------------- kernel 2: xs = silu(causal_conv(xc) + b_conv) ----------------
__global__ void k_conv_silu(const float* __restrict__ xz, const float* __restrict__ Wc,
                            const float* __restrict__ bc,
                            float* __restrict__ xs, _Float16* __restrict__ xs16)
{
    long i = (long)blockIdx.x * blockDim.x + threadIdx.x;
    if (i >= (long)MROWS * DI) return;
    int  d = (int)(i % DI);
    long m = i / DI;
    int  l = (int)(m % LL);
    float acc = bc[d];
    #pragma unroll
    for (int k = 0; k < KC; ++k) {
        int ls = l + k - (KC - 1);
        if (ls >= 0)
            acc += xz[(m + (long)(ls - l)) * NXZ + d] * Wc[d * KC + k];
    }
    float s = 1.f / (1.f + __expf(-acc));
    float v = acc * s;
    xs[i]   = v;
    xs16[i] = (_Float16)v;
}

// ---------------- kernel 3: dbc = xs @ W_xproj  (4096 x 512 x 64) ----------------
__global__ void k_gemm_dbc(const _Float16* __restrict__ xs16, const _Float16* __restrict__ WxT,
                           float* __restrict__ dbc, _Float16* __restrict__ dbc16)
{
    const int m0 = blockIdx.x * 16;              // single 64-col group

    v8f acc = wmma_gemm_16x64_f16(xs16 + (long)m0 * DI, DI, WxT, DI, DI);

    const int lane = threadIdx.x & 31;
    const int wv   = threadIdx.x >> 5;
    const int half = lane >> 4, col = lane & 15;
    const int nbase = wv * 16 + col;
    #pragma unroll
    for (int r = 0; r < 8; ++r) {
        int mr = m0 + r + 8 * half;
        dbc  [(long)mr * NDBC + nbase] = acc[r];
        dbc16[(long)mr * NDBC + nbase] = (_Float16)acc[r];
    }
}

// ---------------- kernel 4: dt = softplus(dbc[:, :32] @ W_dt + b_dt) ----------------
// W_dt^T is zero-padded to K=64, so A can read the full 64-col dbc16 rows.
__global__ void k_gemm_dt(const _Float16* __restrict__ dbc16, const _Float16* __restrict__ WdtT,
                          const float* __restrict__ bdt, float* __restrict__ dt)
{
    const int tiles_m = MROWS / 16;
    const int tm = blockIdx.x % tiles_m;
    const int tg = blockIdx.x / tiles_m;         // 0..7
    const int m0 = tm * 16;

    v8f acc = wmma_gemm_16x64_f16(dbc16 + (long)m0 * NDBC, NDBC,
                                  WdtT + (long)tg * 64 * KDT, KDT, KDT);

    const int lane = threadIdx.x & 31;
    const int wv   = threadIdx.x >> 5;
    const int half = lane >> 4, col = lane & 15;
    const int nn = tg * 64 + wv * 16 + col;
    #pragma unroll
    for (int r = 0; r < 8; ++r) {
        int mr = m0 + r + 8 * half;
        float v = acc[r] + bdt[nn];
        float sp = (v > 20.f) ? v : log1pf(__expf(v));
        dt[(long)mr * DI + nn] = sp;
    }
}

// ---------------- kernel 5: selective scan + output gating ----------------
// block = 256 threads = 16 d-channels x 16 states; grid = BB * (DI/16)
// h[t] = exp(dt*A)*h[t-1] + dt*B*xs ;  y = sum_n h*C + D*xs ;  yact = y*silu(z)
__global__ void k_scan(const float* __restrict__ xz, const float* __restrict__ xs,
                       const float* __restrict__ dbc, const float* __restrict__ dt,
                       const float* __restrict__ A_log, const float* __restrict__ Dv,
                       _Float16* __restrict__ yact16)
{
    const int tid = threadIdx.x;
    const int n   = tid & 15;
    const int dl  = tid >> 4;
    const int blk = blockIdx.x;
    const int bbq = blk / (DI / 16);
    const int d   = (blk % (DI / 16)) * 16 + dl;

    const float Ac = -__expf(A_log[d * NS + n]);   // = -(n+1) numerically
    const float Dd = Dv[d];
    float h = 0.f;
    const long mbase = (long)bbq * LL;

    for (int l = 0; l < LL; ++l) {
        const long m = mbase + l;
        const float dtv = dt[m * DI + d];
        const float xsv = xs[m * DI + d];
        const float Bv  = dbc[m * NDBC + RR + n];
        const float Cv  = dbc[m * NDBC + RR + NS + n];
        const float dA  = __expf(dtv * Ac);
        h = dA * h + dtv * Bv * xsv;
        float c = h * Cv;
        // reduce over 16 states (lanes with same d are 16 consecutive lanes)
        c += __shfl_xor(c, 1, 32);
        c += __shfl_xor(c, 2, 32);
        c += __shfl_xor(c, 4, 32);
        c += __shfl_xor(c, 8, 32);
        if (n == 0) {
            float zv = xz[m * NXZ + DI + d];
            float y  = c + Dd * xsv;
            y *= zv / (1.f + __expf(-zv));   // y * silu(z)
            yact16[m * DI + d] = (_Float16)y;
        }
    }
}

// ---------------- kernel 6: out (+)= flip?(yact @ W_out) ----------------
__global__ void k_gemm_out(const _Float16* __restrict__ yact16, const _Float16* __restrict__ WoutT,
                           float* __restrict__ out, int flip, int accum)
{
    const int tiles_m = MROWS / 16;
    const int tm = blockIdx.x % tiles_m;
    const int tg = blockIdx.x / tiles_m;         // 0..7
    const int m0 = tm * 16;

    v8f acc = wmma_gemm_16x64_f16(yact16 + (long)m0 * DI, DI,
                                  WoutT + (long)tg * 64 * DI, DI, DI);

    const int lane = threadIdx.x & 31;
    const int wv   = threadIdx.x >> 5;
    const int half = lane >> 4, col = lane & 15;
    const int nbase = tg * 64 + wv * 16 + col;
    #pragma unroll
    for (int r = 0; r < 8; ++r) {
        int mr = m0 + r + 8 * half;
        int bbq = mr / LL, l = mr % LL;
        long orow = flip ? ((long)bbq * LL + (LL - 1 - l)) : (long)mr;
        float* p = out + orow * DM + nbase;
        if (accum) *p += acc[r];
        else       *p  = acc[r];
    }
}

// -------------------------------------------------------------------------
extern "C" void kernel_launch(void* const* d_in, const int* in_sizes, int n_in,
                              void* d_out, int out_size, void* d_ws, size_t ws_size,
                              hipStream_t stream)
{
    (void)in_sizes; (void)n_in; (void)out_size; (void)ws_size;
    const float* x = (const float*)d_in[0];
    float* out = (float*)d_out;

    // ---- workspace partition (16B-aligned slices) ----
    char* w = (char*)d_ws;
    float* xz    = (float*)w;                   w += (size_t)MROWS * NXZ  * 4;  // 16 MB
    float* xs    = (float*)w;                   w += (size_t)MROWS * DI   * 4;  //  8 MB
    float* dbc   = (float*)w;                   w += (size_t)MROWS * NDBC * 4;  //  1 MB
    float* dtb   = (float*)w;                   w += (size_t)MROWS * DI   * 4;  //  8 MB
    _Float16* x16    = (_Float16*)w;            w += (size_t)MROWS * DM   * 2;  //  4 MB
    _Float16* xs16   = (_Float16*)w;            w += (size_t)MROWS * DI   * 2;  //  4 MB
    _Float16* dbc16  = (_Float16*)w;            w += (size_t)MROWS * NDBC * 2;  // .5 MB
    _Float16* yact16 = (_Float16*)w;            w += (size_t)MROWS * DI   * 2;  //  4 MB
    _Float16* WinT   = (_Float16*)w;            w += (size_t)NXZ * DM * 2;      //  1 MB
    _Float16* WxT    = (_Float16*)w;            w += (size_t)NDBC * DI * 2;
    _Float16* WdtT   = (_Float16*)w;            w += (size_t)DI * KDT * 2;
    _Float16* WoutT  = (_Float16*)w;            w += (size_t)DM * DI * 2;

    // x -> f16 once (direction handled by negative stride)
    k_cvt_x<<< (MROWS * DM / 4 + 255) / 256, 256, 0, stream >>>(x, x16, (long)MROWS * DM / 4);

    for (int dir = 0; dir < 2; ++dir) {
        const int base = 1 + dir * 9;
        const float* Win   = (const float*)d_in[base + 0];
        const float* Wconv = (const float*)d_in[base + 1];
        const float* bconv = (const float*)d_in[base + 2];
        const float* Wx    = (const float*)d_in[base + 3];
        const float* Wdt   = (const float*)d_in[base + 4];
        const float* bdt   = (const float*)d_in[base + 5];
        const float* Alog  = (const float*)d_in[base + 6];
        const float* Dv    = (const float*)d_in[base + 7];
        const float* Wout  = (const float*)d_in[base + 8];

        // transpose-convert this direction's weights (B operands, [n][k] f16)
        k_cvt_wT<<< ((long)NXZ * DM  + 255) / 256, 256, 0, stream >>>(Win,  WinT,  DM, NXZ, DM);
        k_cvt_wT<<< ((long)NDBC * DI + 255) / 256, 256, 0, stream >>>(Wx,   WxT,   DI, NDBC, DI);
        k_cvt_wT<<< ((long)DI * KDT  + 255) / 256, 256, 0, stream >>>(Wdt,  WdtT,  RR, DI, KDT);
        k_cvt_wT<<< ((long)DM * DI   + 255) / 256, 256, 0, stream >>>(Wout, WoutT, DI, DM, DI);

        k_gemm_xz <<< (MROWS / 16) * (NXZ / 64), 128, 0, stream >>>(x16, WinT, xz, dir);
        k_conv_silu<<< (MROWS * DI) / 256, 256, 0, stream >>>(xz, Wconv, bconv, xs, xs16);
        k_gemm_dbc<<< (MROWS / 16), 128, 0, stream >>>(xs16, WxT, dbc, dbc16);
        k_gemm_dt <<< (MROWS / 16) * (DI / 64), 128, 0, stream >>>(dbc16, WdtT, bdt, dtb);
        k_scan    <<< BB * (DI / 16), 256, 0, stream >>>(xz, xs, dbc, dtb, Alog, Dv, yact16);
        k_gemm_out<<< (MROWS / 16) * (DM / 64), 128, 0, stream >>>(yact16, WoutT, out, dir, dir);
    }
}
